// ASDTransformer_54451595378800
// MI455X (gfx1250) — compile-verified, matrix operand order checked by reading, not run on hardware
//
#include <hip/hip_runtime.h>
#include <hip/hip_bf16.h>
#include <math.h>

// ---------------------------------------------------------------------------
// Model constants
// ---------------------------------------------------------------------------
#define BB    512
#define TT    150
#define DD    256
#define HH    8
#define HD    32
#define DEPTH 6
#define FF    1024
#define NC    4
#define SREAL 151          // T + 1 (CLS)
#define SPAD  160          // padded sequence (multiple of 16)
#define MROWS (BB * SPAD)  // 81920, multiple of 32

typedef __attribute__((ext_vector_type(8)))  _Float16 v8h;
typedef __attribute__((ext_vector_type(16))) _Float16 v16h;
typedef __attribute__((ext_vector_type(8)))  float    v8f;

// Load one 16x32 (A) or 32x16 (B, stored transposed N x K) f16 fragment.
// Per-lane: two contiguous 16-byte chunks at k_base and k_base+16.
__device__ __forceinline__ v16h load_frag(const _Float16* p) {
    v8h lo = *reinterpret_cast<const v8h*>(p);
    v8h hi = *reinterpret_cast<const v8h*>(p + 16);
    v16h r;
#pragma unroll
    for (int i = 0; i < 8; ++i) { r[i] = lo[i]; r[i + 8] = hi[i]; }
    return r;
}

__device__ __forceinline__ v8f wmma_f16(v16h a, v16h b, v8f c) {
    return __builtin_amdgcn_wmma_f32_16x16x32_f16(false, a, false, b,
                                                  (short)0, c, false, false);
}

// ---------------------------------------------------------------------------
// Weight convert: W (K x N, f32) -> Wt (N x K, f16)
// ---------------------------------------------------------------------------
__global__ void convert_w_kernel(const float* __restrict__ W,
                                 _Float16* __restrict__ Wt, int K, int N) {
    int idx = blockIdx.x * blockDim.x + threadIdx.x;
    if (idx >= K * N) return;
    int k = idx / N, n = idx % N;
    Wt[(size_t)n * K + k] = (_Float16)W[idx];
}

// ---------------------------------------------------------------------------
// Embedding: build x (MROWS x D) f32 + f16, padded rows zeroed
// ---------------------------------------------------------------------------
__global__ void embed_kernel(const int* __restrict__ token_types,
                             const int* __restrict__ decision_ids,
                             const float* __restrict__ action_vals,
                             const float* __restrict__ dis_vals,
                             const float* __restrict__ v_vals,
                             const float* __restrict__ action_W,
                             const float* __restrict__ action_b,
                             const float* __restrict__ dis_W,
                             const float* __restrict__ dis_b,
                             const float* __restrict__ v_W,
                             const float* __restrict__ v_b,
                             const float* __restrict__ table,
                             const float* __restrict__ pos,
                             const float* __restrict__ cls,
                             float* __restrict__ xf,
                             _Float16* __restrict__ xh) {
    int row = blockIdx.x;              // b * SPAD + s
    int b = row / SPAD, s = row % SPAD;
    int d = threadIdx.x;
    float val = 0.0f;
    if (s == 0) {
        val = cls[d] + pos[d];
    } else if (s < SREAL) {
        int t = s - 1;
        int idx = b * TT + t;
        int type = token_types[idx];
        if (type == 0) {
            val = action_vals[idx] * action_W[d] + action_b[d];
        } else if (type == 1) {
            float acc = dis_b[d] + v_b[d];
#pragma unroll
            for (int i = 0; i < 10; ++i) acc += dis_vals[idx * 10 + i] * dis_W[i * DD + d];
#pragma unroll
            for (int i = 0; i < 3; ++i) acc += v_vals[idx * 3 + i] * v_W[i * DD + d];
            val = acc;
        } else if (type == 2) {
            val = table[decision_ids[idx] * DD + d];
        } else {
            val = 0.0f;
        }
        val += pos[s * DD + d];
    }
    xf[(size_t)row * DD + d] = val;
    xh[(size_t)row * DD + d] = (_Float16)val;
}

// ---------------------------------------------------------------------------
// Generic WMMA GEMM: C = act(A(MxK,f16) * Bt(NxK,f16)^T + bias) [+ resid]
// Block = 128 threads (4 waves). Wave tile = 32 rows x 64 cols.
// grid.x = (M/32) * (N/256).  K must be a multiple of 64.
// Ping-pong double buffer (no loop-carried register copies).
// ---------------------------------------------------------------------------
struct FragPtrs {
    const _Float16* a0;
    const _Float16* a1;
    const _Float16* b0;
    const _Float16* b1;
    const _Float16* b2;
    const _Float16* b3;
};

__device__ __forceinline__ void load_set(const FragPtrs& fp, int d,
                                         v16h af[2], v16h bf[4]) {
    af[0] = load_frag(fp.a0 + d);
    af[1] = load_frag(fp.a1 + d);
    bf[0] = load_frag(fp.b0 + d);
    bf[1] = load_frag(fp.b1 + d);
    bf[2] = load_frag(fp.b2 + d);
    bf[3] = load_frag(fp.b3 + d);
}

__device__ __forceinline__ void mma_set(v8f acc[2][4], const v16h af[2],
                                        const v16h bf[4]) {
#pragma unroll
    for (int mi = 0; mi < 2; ++mi)
#pragma unroll
        for (int ni = 0; ni < 4; ++ni)
            acc[mi][ni] = wmma_f16(af[mi], bf[ni], acc[mi][ni]);
}

__global__ __launch_bounds__(128, 1)
void gemm_wmma_kernel(const _Float16* __restrict__ A,
                      const _Float16* __restrict__ Bt,
                      const float* __restrict__ bias,
                      const float* __restrict__ resid,
                      float* __restrict__ Cf,
                      _Float16* __restrict__ Ch,
                      int M, int N, int K, int relu) {
    const int strips = N >> 8;                 // N / 256
    const int mstrip = blockIdx.x / strips;
    const int nstrip = blockIdx.x % strips;
    const int wave = threadIdx.x >> 5;
    const int lane = threadIdx.x & 31;
    const int lrow = lane & 15;
    const int lhalf = lane >> 4;

    const int m0 = mstrip * 32;
    const int n0 = nstrip * 256 + wave * 64;

    // Per-lane base pointers (k offset folded in)
    const int kb0 = lhalf * 8;
    FragPtrs fp;
    fp.a0 = A + (size_t)(m0 + lrow) * K + kb0;
    fp.a1 = A + (size_t)(m0 + 16 + lrow) * K + kb0;
    fp.b0 = Bt + (size_t)(n0 + lrow) * K + kb0;
    fp.b1 = Bt + (size_t)(n0 + 16 + lrow) * K + kb0;
    fp.b2 = Bt + (size_t)(n0 + 32 + lrow) * K + kb0;
    fp.b3 = Bt + (size_t)(n0 + 48 + lrow) * K + kb0;

    v8f acc[2][4] = {};

    // Ping-pong fragment sets; each outer iteration consumes 64 K-elements.
    v16h aA[2], bA[4], aB[2], bB[4];
    load_set(fp, 0, aA, bA);

#pragma unroll 1
    for (int kk = 0; kk < K; kk += 64) {
        load_set(fp, kk + 32, aB, bB);     // always in range (kk <= K-64)
        mma_set(acc, aA, bA);
        if (kk + 64 < K) load_set(fp, kk + 64, aA, bA);
        mma_set(acc, aB, bB);
    }

    // Epilogue. D layout: VGPR r -> row = r + 8*lhalf, col = lrow (per 16x16 tile)
#pragma unroll
    for (int mi = 0; mi < 2; ++mi) {
#pragma unroll
        for (int ni = 0; ni < 4; ++ni) {
            const int col = n0 + ni * 16 + lrow;
            const float bv = bias ? bias[col] : 0.0f;
#pragma unroll
            for (int r = 0; r < 8; ++r) {
                const int row = m0 + mi * 16 + r + 8 * lhalf;
                float v = acc[mi][ni][r] + bv;
                if (relu) v = v > 0.0f ? v : 0.0f;
                if (resid) v += resid[(size_t)row * N + col];
                if (Cf) Cf[(size_t)row * N + col] = v;
                if (Ch) Ch[(size_t)row * N + col] = (_Float16)v;
            }
        }
    }
}

// ---------------------------------------------------------------------------
// Attention: one block per (b, h), 10 waves; each wave owns 16 query rows.
// qkvh layout: row = b*SPAD + s (768 halfs): [q | k | v], head slice h*32.
// ---------------------------------------------------------------------------
__global__ __launch_bounds__(320, 1)
void attention_kernel(const _Float16* __restrict__ qkvh,
                      _Float16* __restrict__ oh) {
    const int b = blockIdx.x;
    const int h = blockIdx.y;

    __shared__ __align__(16) _Float16 Vt[HD * SPAD];         // V transposed (hd x key)
    __shared__ __align__(16) _Float16 P[10][16 * SPAD];      // probs, per wave

    // Stage V^T into LDS (all 320 threads)
    for (int e = threadIdx.x; e < HD * SPAD; e += blockDim.x) {
        const int hd = e / SPAD, j = e % SPAD;
        Vt[e] = qkvh[((size_t)(b * SPAD + j)) * 768 + 512 + h * HD + hd];
    }
    __syncthreads();

    const int wave = threadIdx.x >> 5;
    const int lane = threadIdx.x & 31;
    const int lrow = lane & 15;
    const int lhalf = lane >> 4;
    const int kb = lhalf * 8;
    const int q0 = wave * 16;

    // Q fragment (16 queries x 32 dims)
    const _Float16* qp = qkvh + ((size_t)(b * SPAD + q0 + lrow)) * 768 + h * HD + kb;
    v16h aq = load_frag(qp);

    // Scores: 10 tiles of 16 keys, single K-chunk (HD == 32)
    v8f sacc[10];
#pragma unroll
    for (int nt = 0; nt < 10; ++nt) {
        const _Float16* kp =
            qkvh + ((size_t)(b * SPAD + nt * 16 + lrow)) * 768 + 256 + h * HD + kb;
        v16h bk = load_frag(kp);
        v8f z = {};
        sacc[nt] = wmma_f16(aq, bk, z);
    }

    // Softmax per query row. Row data lives across nt (10 vals) and the 16
    // lanes of this half-wave (lanes 0-15 / 16-31 hold distinct query rows).
    const float scale = 0.176776695296636881f; // 1/sqrt(32)
    const int jlane = lrow;
#pragma unroll
    for (int r = 0; r < 8; ++r) {
        float mx = -3.0e38f;
#pragma unroll
        for (int nt = 0; nt < 10; ++nt) {
            const int j = nt * 16 + jlane;
            float s = sacc[nt][r] * scale;
            if (j >= SREAL) s = -3.0e38f;
            sacc[nt][r] = s;
            mx = fmaxf(mx, s);
        }
#pragma unroll
        for (int off = 8; off > 0; off >>= 1)
            mx = fmaxf(mx, __shfl_xor(mx, off, 16));
        float sum = 0.0f;
#pragma unroll
        for (int nt = 0; nt < 10; ++nt) {
            float e = __expf(sacc[nt][r] - mx);
            sacc[nt][r] = e;
            sum += e;
        }
#pragma unroll
        for (int off = 8; off > 0; off >>= 1)
            sum += __shfl_xor(sum, off, 16);
        const float inv = 1.0f / sum;
        const int m = r + 8 * lhalf;
#pragma unroll
        for (int nt = 0; nt < 10; ++nt)
            P[wave][m * SPAD + nt * 16 + jlane] = (_Float16)(sacc[nt][r] * inv);
    }

    // O (16 x 32) = P (16 x 160) @ V (160 x 32): 5 K-chunks x 2 N tiles
    v8f oacc[2] = {};
#pragma unroll
    for (int c = 0; c < 5; ++c) {
        const int kk = c * 32;
        v16h ap = load_frag(&P[wave][lrow * SPAD + kk + kb]);
#pragma unroll
        for (int ni = 0; ni < 2; ++ni) {
            v16h bv = load_frag(&Vt[(ni * 16 + lrow) * SPAD + kk + kb]);
            oacc[ni] = wmma_f16(ap, bv, oacc[ni]);
        }
    }

#pragma unroll
    for (int ni = 0; ni < 2; ++ni) {
#pragma unroll
        for (int r = 0; r < 8; ++r) {
            const int m = r + 8 * lhalf;
            const int hd = ni * 16 + lrow;
            oh[((size_t)(b * SPAD + q0 + m)) * DD + h * HD + hd] = (_Float16)oacc[ni][r];
        }
    }
}

// ---------------------------------------------------------------------------
// LayerNorm over D=256 per row; one wave per row, 8 elems per lane.
// ---------------------------------------------------------------------------
__global__ void layernorm_kernel(const float* __restrict__ in,
                                 const float* __restrict__ g,
                                 const float* __restrict__ bta,
                                 float* __restrict__ outF,
                                 _Float16* __restrict__ outH) {
    const int row = blockIdx.x * (blockDim.x >> 5) + (threadIdx.x >> 5);
    const int lane = threadIdx.x & 31;
    const float* p = in + (size_t)row * DD + lane * 8;
    float v[8];
    float4 a0 = *reinterpret_cast<const float4*>(p);
    float4 a1 = *reinterpret_cast<const float4*>(p + 4);
    v[0] = a0.x; v[1] = a0.y; v[2] = a0.z; v[3] = a0.w;
    v[4] = a1.x; v[5] = a1.y; v[6] = a1.z; v[7] = a1.w;

    float s = 0.0f;
#pragma unroll
    for (int i = 0; i < 8; ++i) s += v[i];
#pragma unroll
    for (int off = 16; off > 0; off >>= 1) s += __shfl_xor(s, off, 32);
    const float mean = s * (1.0f / DD);

    float vs = 0.0f;
#pragma unroll
    for (int i = 0; i < 8; ++i) { const float d = v[i] - mean; vs += d * d; }
#pragma unroll
    for (int off = 16; off > 0; off >>= 1) vs += __shfl_xor(vs, off, 32);
    const float rstd = rsqrtf(vs * (1.0f / DD) + 1e-5f);

#pragma unroll
    for (int i = 0; i < 8; ++i) {
        const int d = lane * 8 + i;
        const float o = (v[i] - mean) * rstd * g[d] + bta[d];
        outF[(size_t)row * DD + d] = o;
        outH[(size_t)row * DD + d] = (_Float16)o;
    }
}

// ---------------------------------------------------------------------------
// Heads: per batch (one wave): cls = x[b, 0, :]; two LNs + tiny GEMVs.
// out layout: [decision_logits (512*4) | action_pred (512)]
// ---------------------------------------------------------------------------
__global__ void heads_kernel(const float* __restrict__ xf,
                             const float* __restrict__ dec_g,
                             const float* __restrict__ dec_bn,
                             const float* __restrict__ decW,
                             const float* __restrict__ decB,
                             const float* __restrict__ act_g,
                             const float* __restrict__ act_bn,
                             const float* __restrict__ actW,
                             const float* __restrict__ actB,
                             float* __restrict__ out) {
    const int b = blockIdx.x;
    const int lane = threadIdx.x;
    const float* p = xf + (size_t)(b * SPAD) * DD + lane * 8;
    float v[8];
    float4 a0 = *reinterpret_cast<const float4*>(p);
    float4 a1 = *reinterpret_cast<const float4*>(p + 4);
    v[0] = a0.x; v[1] = a0.y; v[2] = a0.z; v[3] = a0.w;
    v[4] = a1.x; v[5] = a1.y; v[6] = a1.z; v[7] = a1.w;

    float s = 0.0f;
#pragma unroll
    for (int i = 0; i < 8; ++i) s += v[i];
#pragma unroll
    for (int off = 16; off > 0; off >>= 1) s += __shfl_xor(s, off, 32);
    const float mean = s * (1.0f / DD);
    float vs = 0.0f;
#pragma unroll
    for (int i = 0; i < 8; ++i) { const float d = v[i] - mean; vs += d * d; }
#pragma unroll
    for (int off = 16; off > 0; off >>= 1) vs += __shfl_xor(vs, off, 32);
    const float rstd = rsqrtf(vs * (1.0f / DD) + 1e-5f);

    float pc[NC] = {0.0f, 0.0f, 0.0f, 0.0f};
    float pa = 0.0f;
#pragma unroll
    for (int i = 0; i < 8; ++i) {
        const int d = lane * 8 + i;
        const float xn = (v[i] - mean) * rstd;
        const float xd = xn * dec_g[d] + dec_bn[d];
        const float xa = xn * act_g[d] + act_bn[d];
#pragma unroll
        for (int c = 0; c < NC; ++c) pc[c] += xd * decW[d * NC + c];
        pa += xa * actW[d];
    }
#pragma unroll
    for (int off = 16; off > 0; off >>= 1) {
#pragma unroll
        for (int c = 0; c < NC; ++c) pc[c] += __shfl_xor(pc[c], off, 32);
        pa += __shfl_xor(pa, off, 32);
    }
    if (lane == 0) {
#pragma unroll
        for (int c = 0; c < NC; ++c) out[b * NC + c] = pc[c] + decB[c];
        out[BB * NC + b] = pa + actB[0];
    }
}

// ---------------------------------------------------------------------------
// Launch
// ---------------------------------------------------------------------------
extern "C" void kernel_launch(void* const* d_in, const int* in_sizes, int n_in,
                              void* d_out, int out_size, void* d_ws, size_t ws_size,
                              hipStream_t stream) {
    // Inputs (setup_inputs order)
    const int*   token_types  = (const int*)d_in[0];
    const int*   decision_ids = (const int*)d_in[1];
    const float* action_vals  = (const float*)d_in[2];
    const float* dis_vals     = (const float*)d_in[3];
    const float* v_vals       = (const float*)d_in[4];
    const float* action_W     = (const float*)d_in[5];
    const float* action_b     = (const float*)d_in[6];
    const float* dis_W        = (const float*)d_in[7];
    const float* dis_b        = (const float*)d_in[8];
    const float* v_W          = (const float*)d_in[9];
    const float* v_b          = (const float*)d_in[10];
    const float* table        = (const float*)d_in[11];
    const float* pos_embed    = (const float*)d_in[12];
    const float* cls_token    = (const float*)d_in[13];
    const float* qkv_W        = (const float*)d_in[14];
    const float* qkv_b        = (const float*)d_in[15];
    const float* out_W        = (const float*)d_in[16];
    const float* out_b        = (const float*)d_in[17];
    const float* ln1_g        = (const float*)d_in[18];
    const float* ln1_b        = (const float*)d_in[19];
    const float* ln2_g        = (const float*)d_in[20];
    const float* ln2_b        = (const float*)d_in[21];
    const float* ffn1_W       = (const float*)d_in[22];
    const float* ffn1_b       = (const float*)d_in[23];
    const float* ffn2_W       = (const float*)d_in[24];
    const float* ffn2_b       = (const float*)d_in[25];
    const float* dec_ln_g     = (const float*)d_in[26];
    const float* dec_ln_b     = (const float*)d_in[27];
    const float* dec_W        = (const float*)d_in[28];
    const float* dec_b        = (const float*)d_in[29];
    const float* act_ln_g     = (const float*)d_in[30];
    const float* act_ln_b     = (const float*)d_in[31];
    const float* act_W        = (const float*)d_in[32];
    const float* act_b        = (const float*)d_in[33];

    // Workspace layout
    char* ws = (char*)d_ws;
    const size_t SZ_XF  = (size_t)MROWS * DD * 4;   // 83,886,080
    const size_t SZ_XH  = (size_t)MROWS * DD * 2;   // 41,943,040
    const size_t SZ_BIG = (size_t)MROWS * FF * 2;   // 167,772,160 (qkvh & h1h share)
    float*    xf   = (float*)ws;
    _Float16* xh   = (_Float16*)(ws + SZ_XF);
    float*    tmp  = (float*)(ws + SZ_XF + SZ_XH);
    _Float16* big  = (_Float16*)(ws + 2 * SZ_XF + SZ_XH);
    _Float16* oh   = (_Float16*)(ws + 2 * SZ_XF + SZ_XH + SZ_BIG);
    _Float16* wts  = (_Float16*)(ws + 2 * SZ_XF + 2 * SZ_XH + SZ_BIG);
    _Float16* qkvWt  = wts;                        // 6 * 768*256
    _Float16* outWt  = qkvWt + 6 * 768 * 256;      // 6 * 256*256
    _Float16* ffn1Wt = outWt + 6 * 256 * 256;      // 6 * 1024*256
    _Float16* ffn2Wt = ffn1Wt + 6 * 1024 * 256;    // 6 * 256*1024
    _Float16* qkvh = big;  // M x 768
    _Float16* h1h  = big;  // M x 1024 (disjoint in time from qkvh)

    // 1. Convert + transpose weights to f16
    for (int i = 0; i < DEPTH; ++i) {
        convert_w_kernel<<<(256 * 768 + 255) / 256, 256, 0, stream>>>(
            qkv_W + (size_t)i * 256 * 768, qkvWt + (size_t)i * 768 * 256, 256, 768);
        convert_w_kernel<<<(256 * 256 + 255) / 256, 256, 0, stream>>>(
            out_W + (size_t)i * 256 * 256, outWt + (size_t)i * 256 * 256, 256, 256);
        convert_w_kernel<<<(256 * 1024 + 255) / 256, 256, 0, stream>>>(
            ffn1_W + (size_t)i * 256 * 1024, ffn1Wt + (size_t)i * 1024 * 256, 256, 1024);
        convert_w_kernel<<<(1024 * 256 + 255) / 256, 256, 0, stream>>>(
            ffn2_W + (size_t)i * 1024 * 256, ffn2Wt + (size_t)i * 256 * 1024, 1024, 256);
    }

    // 2. Embedding
    embed_kernel<<<MROWS, DD, 0, stream>>>(
        token_types, decision_ids, action_vals, dis_vals, v_vals,
        action_W, action_b, dis_W, dis_b, v_W, v_b,
        table, pos_embed, cls_token, xf, xh);

    const int mstrips = MROWS / 32;  // 2560

    // 3. Transformer layers
    for (int i = 0; i < DEPTH; ++i) {
        // QKV: (M x 256) @ (256 x 768) + b  -> qkvh (f16)
        gemm_wmma_kernel<<<mstrips * (768 / 256), 128, 0, stream>>>(
            xh, qkvWt + (size_t)i * 768 * 256, qkv_b + (size_t)i * 768,
            nullptr, nullptr, qkvh, MROWS, 768, 256, 0);

        // Attention per (b, h)
        attention_kernel<<<dim3(BB, HH), 320, 0, stream>>>(qkvh, oh);

        // Out proj + residual -> tmp (f32)
        gemm_wmma_kernel<<<mstrips, 128, 0, stream>>>(
            oh, outWt + (size_t)i * 256 * 256, out_b + (size_t)i * 256,
            xf, tmp, nullptr, MROWS, 256, 256, 0);

        // LN1 -> xf, xh
        layernorm_kernel<<<MROWS / 8, 256, 0, stream>>>(
            tmp, ln1_g + (size_t)i * 256, ln1_b + (size_t)i * 256, xf, xh);

        // FFN1 + ReLU -> h1h (f16)
        gemm_wmma_kernel<<<mstrips * (1024 / 256), 128, 0, stream>>>(
            xh, ffn1Wt + (size_t)i * 1024 * 256, ffn1_b + (size_t)i * 1024,
            nullptr, nullptr, h1h, MROWS, 1024, 256, 1);

        // FFN2 + residual -> tmp (f32)
        gemm_wmma_kernel<<<mstrips, 128, 0, stream>>>(
            h1h, ffn2Wt + (size_t)i * 256 * 1024, ffn2_b + (size_t)i * 256,
            xf, tmp, nullptr, MROWS, 256, 1024, 0);

        // LN2 -> xf, xh
        layernorm_kernel<<<MROWS / 8, 256, 0, stream>>>(
            tmp, ln2_g + (size_t)i * 256, ln2_b + (size_t)i * 256, xf, xh);
    }

    // 4. Heads
    heads_kernel<<<BB, 32, 0, stream>>>(
        xf, dec_ln_g, dec_ln_b, dec_W, dec_b,
        act_ln_g, act_ln_b, act_W, act_b, (float*)d_out);
}
